// MambaRegressor_72567767433403
// MI455X (gfx1250) — matrix operand order repaired and checked by reading, not compiled, not run on hardware
//
#include <hip/hip_runtime.h>
#include <hip/hip_bf16.h>
#include <math.h>

typedef __attribute__((ext_vector_type(16))) __bf16 v16bf;
typedef __attribute__((ext_vector_type(8)))  __bf16 v8bf;
typedef __attribute__((ext_vector_type(8)))  float  v8f;

#define B_  8
#define L_  2048
#define H_  256
#define NL_ 4
#define N_  16
#define I_  512          // 2*H
#define R_  16           // H/16
#define K_  4
#define ROWS (B_*L_)     // 16384
#define EPS_ 1e-5f

// LDS row stride (elements) for staged panels: 32 data + 8 pad -> 80B rows,
// 16B aligned, 20-word stride -> conflict-free across 16 lanes (64 banks).
#define LDSP 40
#define SA_BYTES (64  * LDSP * 2)   // 5120B per buffer
#define SB_BYTES (128 * LDSP * 2)   // 10240B per buffer

// -------------------------- CDNA5 async global->LDS copy (ASYNCcnt tracked)
__device__ __forceinline__ void async_g2l_b128(unsigned lds_off, const void* g) {
  asm volatile("global_load_async_to_lds_b128 %0, %1, off"
               :: "v"(lds_off), "v"((unsigned long long)(size_t)g)
               : "memory");
}
__device__ __forceinline__ void wait_async0() {
  asm volatile("s_wait_asynccnt 0" ::: "memory");
}

__device__ __forceinline__ v8f wmma_bf16(v16bf a, v16bf b, v8f c) {
  return __builtin_amdgcn_wmma_f32_16x16x32_bf16(false, a, false, b,
                                                 (short)0, c, false, false);
}
// A 16x32 frag: lanes 0-15 K=0..7 (v0-3) & 16..23 (v4-7); lanes 16-31 +8
__device__ __forceinline__ v16bf ld_fragA(const __bf16* p, int kh) {
  v8bf lo = *(const v8bf*)(p + 8 * kh);
  v8bf hi = *(const v8bf*)(p + 16 + 8 * kh);
  v16bf f;
#pragma unroll
  for (int j = 0; j < 8; ++j) { f[j] = lo[j]; f[j + 8] = hi[j]; }
  return f;
}
// B 32x16 frag: lane column n, K = 8*kh..8*kh+7 & 16+8*kh..16+8*kh+7
__device__ __forceinline__ v16bf ld_fragB(const __bf16* p, int kh) {
  v8bf lo = *(const v8bf*)(p + 8 * kh);
  v8bf hi = *(const v8bf*)(p + 16 + 8 * kh);
  v16bf f;
#pragma unroll
  for (int j = 0; j < 8; ++j) { f[j] = lo[j]; f[j + 8] = hi[j]; }
  return f;
}

// ---------------------------------------------------------------- embed gather
__global__ void k_embed(const int* __restrict__ ids, const float* __restrict__ emb,
                        float* __restrict__ x) {
  int tid = blockIdx.x * blockDim.x + threadIdx.x;     // ROWS*H_
  int col = tid & (H_ - 1);
  int row = tid >> 8;                                  // H_ == 256
  x[tid] = emb[(size_t)ids[row] * H_ + col];
}

// ------------------------------------------------------- rmsnorm -> bf16 row
__global__ void k_rmsnorm_bf16(const float* __restrict__ x, const float* __restrict__ w,
                               __bf16* __restrict__ out) {
  __shared__ float red[256];
  int row = blockIdx.x;
  int t   = threadIdx.x;
  float v = x[(size_t)row * H_ + t];
  red[t] = v * v;
  __syncthreads();
  for (int s = 128; s > 0; s >>= 1) {
    if (t < s) red[t] += red[t + s];
    __syncthreads();
  }
  float scale = rsqrtf(red[0] * (1.0f / H_) + EPS_);
  out[(size_t)row * H_ + t] = (__bf16)(v * scale * w[t]);
}

// ------------------------------------------------------------- f32 -> bf16
__global__ void k_cvt_bf16(const float* __restrict__ in, __bf16* __restrict__ out, int n) {
  int i = blockIdx.x * blockDim.x + threadIdx.x;
  if (i < n) out[i] = (__bf16)in[i];
}

// ---------------------------------------------------------------- WMMA GEMM
// C(M,N) = A(M,K) * Bw(N,K)^T ; A,Bw bf16 row-major.
// Double-buffered 64x128 block tile in LDS filled by async global->LDS; each
// wave computes a 32x32 register tile (4 WMMAs / k-step, 2x A + 2x B reuse).
// MODE: 0 = f32 store, 1 = bf16 store, 2 = f32 residual add. Uniform barriers.
template <int MODE, typename OutT>
__global__ void __launch_bounds__(256)
k_gemm_bf16(const __bf16* __restrict__ A, const __bf16* __restrict__ Bw,
            int M, int N, int K, OutT* __restrict__ C) {
  __shared__ __align__(16) __bf16 sA[2][64 * LDSP];
  __shared__ __align__(16) __bf16 sB[2][128 * LDSP];

  const int tid  = threadIdx.x;
  const int wave = tid >> 5, lane = tid & 31;
  const int wm = wave >> 2, wn = wave & 3;             // 2 x 4 wave grid
  const int kh = lane >> 4, r15 = lane & 15;
  const int mblk = blockIdx.y * 64;
  const int nblk = blockIdx.x * 128;

  // copy assignment: one 16B chunk of A, two of B per thread per k-step
  const int c_row = tid >> 2;                          // 0..63
  const int c_off = (tid & 3) * 8;                     // k element offset 0/8/16/24
  int gbr0 = nblk + c_row;       if (gbr0 >= N) gbr0 = N - 1;   // clamp (N=48 case)
  int gbr1 = nblk + 64 + c_row;  if (gbr1 >= N) gbr1 = N - 1;

  // wave-uniform subtile guards, forced into SGPRs (EXEC untouched around WMMA)
  const int n_ok0 = __builtin_amdgcn_readfirstlane((int)((nblk + wn * 32 + 16) <= N));
  const int n_ok1 = __builtin_amdgcn_readfirstlane((int)((nblk + wn * 32 + 32) <= N));

  const unsigned sA_off  = (unsigned)(size_t)&sA[0][c_row * LDSP + c_off];
  const unsigned sB_off0 = (unsigned)(size_t)&sB[0][c_row * LDSP + c_off];
  const unsigned sB_off1 = (unsigned)(size_t)&sB[0][(64 + c_row) * LDSP + c_off];
  const __bf16* gA = A  + (size_t)(mblk + c_row) * K + c_off;
  const __bf16* gB0 = Bw + (size_t)gbr0 * K + c_off;
  const __bf16* gB1 = Bw + (size_t)gbr1 * K + c_off;

  v8f acc00 = {}, acc01 = {}, acc10 = {}, acc11 = {};

  // prologue: stage k=0 into buffer 0
  async_g2l_b128(sA_off,  gA);
  async_g2l_b128(sB_off0, gB0);
  async_g2l_b128(sB_off1, gB1);
  wait_async0();
  __syncthreads();

  int buf = 0;
  for (int k0 = 0; k0 < K; k0 += 32) {
    int kn = k0 + 32;
    if (kn < K) {                                      // prefetch next panel (async)
      unsigned boff = (buf ^ 1) ? 1u : 0u;
      async_g2l_b128(sA_off  + boff * SA_BYTES, gA  + kn);
      async_g2l_b128(sB_off0 + boff * SB_BYTES, gB0 + kn);
      async_g2l_b128(sB_off1 + boff * SB_BYTES, gB1 + kn);
    }

    const __bf16* a0p = &sA[buf][(wm * 32 + r15) * LDSP];
    const __bf16* a1p = &sA[buf][(wm * 32 + 16 + r15) * LDSP];
    const __bf16* b0p = &sB[buf][(wn * 32 + r15) * LDSP];
    const __bf16* b1p = &sB[buf][(wn * 32 + 16 + r15) * LDSP];
    v16bf a0 = ld_fragA(a0p, kh);
    v16bf a1 = ld_fragA(a1p, kh);
    if (n_ok0) {
      v16bf b0 = ld_fragB(b0p, kh);
      acc00 = wmma_bf16(a0, b0, acc00);
      acc10 = wmma_bf16(a1, b0, acc10);
    }
    if (n_ok1) {
      v16bf b1 = ld_fragB(b1p, kh);
      acc01 = wmma_bf16(a0, b1, acc01);
      acc11 = wmma_bf16(a1, b1, acc11);
    }

    wait_async0();                                     // next panel landed
    __syncthreads();                                   // all waves done with `buf`
    buf ^= 1;
  }

  // epilogue: D layout -> lane holds N=n0+r15, rows m0+8*kh .. +7
  const int m0 = mblk + wm * 32;
  const int n0 = nblk + wn * 32;
  auto store_tile = [&](const v8f& acc, int mt, int nt) {
    int ncol  = nt + r15;
    int mbase = mt + 8 * kh;
#pragma unroll
    for (int r = 0; r < 8; ++r) {
      size_t idx = (size_t)(mbase + r) * N + ncol;
      if constexpr (MODE == 0)      C[idx]  = acc[r];
      else if constexpr (MODE == 1) C[idx]  = (__bf16)acc[r];
      else                          C[idx] += acc[r];
    }
  };
  if (n_ok0) { store_tile(acc00, m0, n0);      store_tile(acc10, m0 + 16, n0); }
  if (n_ok1) { store_tile(acc01, m0, n0 + 16); store_tile(acc11, m0 + 16, n0 + 16); }
}

// -------------------------------------- causal depthwise conv (K=4) + SiLU
__global__ void k_conv_silu(const __bf16* __restrict__ proj, const float* __restrict__ cw,
                            const float* __restrict__ cb, float* __restrict__ hs,
                            __bf16* __restrict__ hbf) {
  int tid = blockIdx.x * blockDim.x + threadIdx.x;     // (b,t,i) over ROWS*I_
  int i  = tid & (I_ - 1);
  int bt = tid >> 9;                                   // I_ == 512
  int t  = bt & (L_ - 1);
  float acc = cb[i];
#pragma unroll
  for (int k = 0; k < K_; ++k) {
    int tt = t - (K_ - 1) + k;
    if (tt >= 0)
      acc += (float)proj[(size_t)(bt - t + tt) * (2 * I_) + i] * cw[i * K_ + k];
  }
  float s = acc / (1.0f + __expf(-acc));
  hs[tid]  = s;
  hbf[tid] = (__bf16)s;
}

// ------------------------------------------------ dt = softplus(ssm_dt @ Wdt^T + b)
__global__ void k_dtproj(const float* __restrict__ ssm, const float* __restrict__ w,
                         const float* __restrict__ bias, float* __restrict__ dt) {
  int tid = blockIdx.x * blockDim.x + threadIdx.x;     // ROWS*I_
  int i = tid & (I_ - 1);
  int m = tid >> 9;
  const float* srow = ssm + (size_t)m * 48;
  const float* wrow = w + i * R_;
  float acc = bias[i];
#pragma unroll
  for (int r = 0; r < R_; ++r) acc += srow[r] * wrow[r];
  dt[tid] = (acc > 20.0f) ? acc : log1pf(__expf(acc));
}

// ------------- selective scan: one lane per (b,i,n); 16-lane shuffle reduce.
// Fuses +h*D, silu(gate) gating and bf16 conversion into each step's epilogue.
__global__ void k_scan(const float* __restrict__ hs, const float* __restrict__ dt,
                       const float* __restrict__ ssm, const float* __restrict__ A_log,
                       const float* __restrict__ Dp, const __bf16* __restrict__ proj,
                       __bf16* __restrict__ ygb) {
  int tid = blockIdx.x * blockDim.x + threadIdx.x;     // B_*I_*16 = 65536 threads
  int n = tid & (N_ - 1);
  int i = (tid >> 4) & (I_ - 1);
  int b = tid >> 13;                                   // 4 + 9
  float Ar = -__expf(A_log[i * N_ + n]);
  float s  = 0.0f;
  float Di = Dp[i];
  for (int t = 0; t < L_; ++t) {
    size_t bt = (size_t)b * L_ + t;
    float dtv = dt[bt * I_ + i];                       // broadcast across 16 lanes
    float u   = hs[bt * I_ + i];
    const float* sr = ssm + bt * 48;                   // [16:32)=B [32:48)=C coalesced
    s = __expf(dtv * Ar) * s + (dtv * u) * sr[16 + n];
    float p = s * sr[32 + n];
    p += __shfl_xor(p, 1, 16);
    p += __shfl_xor(p, 2, 16);
    p += __shfl_xor(p, 4, 16);
    p += __shfl_xor(p, 8, 16);
    if (n == 0) {
      float yv = p + u * Di;
      float g  = (float)proj[bt * (2 * I_) + I_ + i];
      float gs = g / (1.0f + __expf(-g));
      ygb[bt * I_ + i] = (__bf16)(yv * gs);
    }
  }
}

// ----------------- final rmsnorm (gathered row only) + GELU MLP head -> (B,)
__global__ void k_head(const float* __restrict__ x, const int* __restrict__ lengths,
                       const float* __restrict__ nw, const float* __restrict__ w1,
                       const float* __restrict__ b1, const float* __restrict__ w2,
                       const float* __restrict__ b2, float* __restrict__ out) {
  __shared__ float red[256];
  __shared__ float xn[256];
  __shared__ float z[64];
  int b = blockIdx.x, t = threadIdx.x;
  int row = lengths[b] - 1;
  if (row < 0) row = 0; if (row >= L_) row = L_ - 1;
  float v = x[((size_t)b * L_ + row) * H_ + t];
  red[t] = v * v;
  __syncthreads();
  for (int s = 128; s > 0; s >>= 1) {
    if (t < s) red[t] += red[t + s];
    __syncthreads();
  }
  float scale = rsqrtf(red[0] * (1.0f / H_) + EPS_);
  xn[t] = v * scale * nw[t];
  __syncthreads();
  if (t < 64) {
    float acc = b1[t];
    for (int k = 0; k < H_; ++k) acc += xn[k] * w1[t * H_ + k];
    z[t] = 0.5f * acc * (1.0f + erff(acc * 0.70710678118654752f));
  }
  __syncthreads();
  if (t == 0) {
    float acc = b2[0];
    for (int j = 0; j < 64; ++j) acc += z[j] * w2[j];
    out[b] = acc;
  }
}

// ================================================================ host side
extern "C" void kernel_launch(void* const* d_in, const int* in_sizes, int n_in,
                              void* d_out, int out_size, void* d_ws, size_t ws_size,
                              hipStream_t stream) {
  (void)in_sizes; (void)n_in; (void)out_size; (void)ws_size;

  // setup_inputs flattening: input_ids, lengths, embed, norm_f_w,
  // 4 x {norm_w,in_proj_w,conv_w,conv_b,x_proj_w,dt_proj_w,dt_proj_b,A_log,D,out_proj_w},
  // head {w1,b1,w2,b2}
  const int*   ids      = (const int*)d_in[0];
  const int*   lengths  = (const int*)d_in[1];
  const float* emb      = (const float*)d_in[2];
  const float* norm_f_w = (const float*)d_in[3];
  const float* hw1 = (const float*)d_in[44];
  const float* hb1 = (const float*)d_in[45];
  const float* hw2 = (const float*)d_in[46];
  const float* hb2 = (const float*)d_in[47];

  char* p = (char*)d_ws;
  auto carve = [&](size_t bytes) -> void* {
    void* r = (void*)p;
    p += (bytes + 255) & ~(size_t)255;
    return r;
  };
  float*  xbuf    = (float*) carve((size_t)ROWS * H_ * 4);        // residual stream
  __bf16* xnbf    = (__bf16*)carve((size_t)ROWS * H_ * 2);        // rmsnorm out
  __bf16* projbf  = (__bf16*)carve((size_t)ROWS * 2 * I_ * 2);    // in_proj out (h|gate)
  float*  hsbuf   = (float*) carve((size_t)ROWS * I_ * 4);        // silu(conv) f32
  __bf16* hbf     = (__bf16*)carve((size_t)ROWS * I_ * 2);        // silu(conv) bf16
  float*  ssmbuf  = (float*) carve((size_t)ROWS * 48 * 4);        // x_proj out
  float*  dtbuf   = (float*) carve((size_t)ROWS * I_ * 4);        // softplus dt
  __bf16* ygb     = (__bf16*)carve((size_t)ROWS * I_ * 2);        // gated scan out
  __bf16* w_in_bf = (__bf16*)carve((size_t)2 * I_ * H_ * 2);
  __bf16* w_x_bf  = (__bf16*)carve((size_t)48 * I_ * 2);
  __bf16* w_out_bf= (__bf16*)carve((size_t)H_ * I_ * 2);

  k_embed<<<(ROWS * H_) / 256, 256, 0, stream>>>(ids, emb, xbuf);

  for (int l = 0; l < NL_; ++l) {
    int base = 4 + l * 10;
    const float* norm_w    = (const float*)d_in[base + 0];
    const float* in_proj_w = (const float*)d_in[base + 1];
    const float* conv_w    = (const float*)d_in[base + 2];
    const float* conv_b    = (const float*)d_in[base + 3];
    const float* x_proj_w  = (const float*)d_in[base + 4];
    const float* dt_proj_w = (const float*)d_in[base + 5];
    const float* dt_proj_b = (const float*)d_in[base + 6];
    const float* A_log     = (const float*)d_in[base + 7];
    const float* Dw        = (const float*)d_in[base + 8];
    const float* out_proj_w= (const float*)d_in[base + 9];

    k_rmsnorm_bf16<<<ROWS, 256, 0, stream>>>(xbuf, norm_w, xnbf);

    // in_proj: (ROWS,256) x (1024,256)^T -> bf16 (ROWS,1024)
    k_cvt_bf16<<<(2 * I_ * H_ + 255) / 256, 256, 0, stream>>>(in_proj_w, w_in_bf, 2 * I_ * H_);
    {
      dim3 g((2 * I_ + 127) / 128, ROWS / 64);
      k_gemm_bf16<1, __bf16><<<g, 256, 0, stream>>>(xnbf, w_in_bf, ROWS, 2 * I_, H_,
                                                    projbf);
    }

    k_conv_silu<<<(ROWS * I_) / 256, 256, 0, stream>>>(projbf, conv_w, conv_b, hsbuf, hbf);

    // x_proj: (ROWS,512) x (48,512)^T -> f32 (ROWS,48)
    k_cvt_bf16<<<(48 * I_ + 255) / 256, 256, 0, stream>>>(x_proj_w, w_x_bf, 48 * I_);
    {
      dim3 g(1, ROWS / 64);
      k_gemm_bf16<0, float><<<g, 256, 0, stream>>>(hbf, w_x_bf, ROWS, 48, I_,
                                                   ssmbuf);
    }

    k_dtproj<<<(ROWS * I_) / 256, 256, 0, stream>>>(ssmbuf, dt_proj_w, dt_proj_b, dtbuf);

    k_scan<<<(B_ * I_ * 16) / 256, 256, 0, stream>>>(hsbuf, dtbuf, ssmbuf, A_log, Dw,
                                                     projbf, ygb);

    // out_proj: (ROWS,512) x (256,512)^T, residual add into xbuf
    k_cvt_bf16<<<(H_ * I_ + 255) / 256, 256, 0, stream>>>(out_proj_w, w_out_bf, H_ * I_);
    {
      dim3 g(H_ / 128, ROWS / 64);
      k_gemm_bf16<2, float><<<g, 256, 0, stream>>>(ygb, w_out_bf, ROWS, H_, I_,
                                                   xbuf);
    }
  }

  k_head<<<B_, 256, 0, stream>>>(xbuf, lengths, norm_f_w, hw1, hb1, hw2, hb2,
                                 (float*)d_out);
}
